// LSTMModel_34754875359360
// MI455X (gfx1250) — matrix operand order repaired and checked by reading, not compiled, not run on hardware
//
#include <hip/hip_runtime.h>
#include <hip/hip_bf16.h>

// ---------------------------------------------------------------------------
// Fused 2-layer LSTM + MLP head for MI455X (gfx1250, wave32, WMMA).
//
// Each wave owns 16 batch rows and runs the full T=128 recurrence privately:
//   gates1 = [x_t | h1] (16x96) @ W1cat(96x256)  -> 48 v_wmma_f32_16x16x32_bf16
//   gates2 = [h1  | h2] (16x96) @ W2cat(96x128)  -> 24 v_wmma_f32_16x16x32_bf16
// Weights staged once into LDS as bf16 (ds_load_b128 fragments); h-state
// round-trips through a private LDS region to convert D-tile layout ->
// A-fragment layout (same-wave LDS is in-order, no barrier needed).
// Activations use v_exp_f32 + single-instruction v_rcp_f32.
// ---------------------------------------------------------------------------

typedef __attribute__((ext_vector_type(16))) __bf16 v16bf;
typedef __attribute__((ext_vector_type(8)))  float  v8f;

#define T_STEPS 128
#define DIN   32
#define H1DIM 64
#define H2DIM 32
#define G1    256   // 4*H1
#define G2    128   // 4*H2
#define K1    96    // DIN + H1
#define K2    96    // H1 + H2
#define WPB   8     // waves per block
#define NTHR  (32*WPB)

#define LOG2E 1.4426950408889634f

// sigmoid(x) = rcp(1 + 2^(-log2e * x)); saturates to 0/1 at extremes.
__device__ __forceinline__ float sigm_f(float x) {
    return __builtin_amdgcn_rcpf(1.0f + __builtin_amdgcn_exp2f(-LOG2E * x));
}
// tanh(x) = 2*sigmoid(2x) - 1; exp2 over/underflow gives exact +/-1 saturation.
__device__ __forceinline__ float tanh_f(float x) {
    float s = __builtin_amdgcn_rcpf(1.0f + __builtin_amdgcn_exp2f(-2.0f * LOG2E * x));
    return __builtin_fmaf(2.0f, s, -1.0f);
}

__device__ __forceinline__ v8f splat8(float v) {
    v8f r;
#pragma unroll
    for (int e = 0; e < 8; ++e) r[e] = v;
    return r;
}

// K offset within a 16-bit A fragment (16x32), per CDNA5 ISA:
//   lanes 0-15 : V0-3 -> K=0..7 (pairs), V4-7 -> K=16..23
//   lanes 16-31: V0-3 -> K=8..15,        V4-7 -> K=24..31
__device__ __forceinline__ int a_kofs(int v, int hi) {
    return (v < 4 ? 2 * v : 8 + 2 * v) + (hi ? 8 : 0);
}

// A fragment (16 x 32 slice starting at kbase) from an LDS row-major bf16
// matrix [16][rowstride]. lane L holds row M = L%16.
__device__ __forceinline__ v16bf load_a_lds(const __bf16* base, int rowstride,
                                            int row, int kbase, int hi) {
    v16bf f;
#pragma unroll
    for (int v = 0; v < 8; ++v) {
        int k = kbase + a_kofs(v, hi);
        f[2 * v]     = base[row * rowstride + k];
        f[2 * v + 1] = base[row * rowstride + k + 1];
    }
    return f;
}

// B fragment (32 x 16): weights stored transposed in LDS as WT[g][k] with row
// stride KS so the (k,k+1) pair is contiguous. Per ISA: lane L holds column
// N = L%16; VGPR v holds K = a_kofs(v, hi) (same K order as the A fragment).
__device__ __forceinline__ v16bf load_b_lds(const __bf16* wt, int KS,
                                            int g, int kbase, int hi) {
    v16bf f;
#pragma unroll
    for (int v = 0; v < 8; ++v) {
        int k = kbase + a_kofs(v, hi);
        f[2 * v]     = wt[g * KS + k];
        f[2 * v + 1] = wt[g * KS + k + 1];
    }
    return f;
}

// A fragment of x_t (16 rows x DIN=32), f32 global -> bf16.
__device__ __forceinline__ v16bf load_a_x(const float* __restrict__ x,
                                          int b0, int t, int row, int hi) {
    const float* p = x + ((size_t)(b0 + row) * T_STEPS + t) * DIN;
    v16bf f;
#pragma unroll
    for (int v = 0; v < 8; ++v) {
        int k = a_kofs(v, hi);
        f[2 * v]     = (__bf16)p[k];
        f[2 * v + 1] = (__bf16)p[k + 1];
    }
    return f;
}

__device__ __forceinline__ v8f wmma_bf16(v16bf a, v16bf b, v8f c) {
    return __builtin_amdgcn_wmma_f32_16x16x32_bf16(
        /*neg_a=*/false, a, /*neg_b=*/false, b,
        /*c_mod=*/(short)0, c, /*reuse_a=*/false, /*reuse_b=*/false);
}

__global__ __launch_bounds__(NTHR)
void lstm_fused_kernel(const float* __restrict__ x,
                       const float* __restrict__ Wih1, const float* __restrict__ Whh1,
                       const float* __restrict__ bih1, const float* __restrict__ bhh1,
                       const float* __restrict__ Wih2, const float* __restrict__ Whh2,
                       const float* __restrict__ bih2, const float* __restrict__ bhh2,
                       const float* __restrict__ fc1w, const float* __restrict__ fc1b,
                       const float* __restrict__ fc2w, const float* __restrict__ fc2b,
                       float* __restrict__ out) {
    extern __shared__ char smem[];
    // LDS layout:
    __bf16* sW1  = (__bf16*)smem;                      // [G1][K1] bf16
    __bf16* sW2  = sW1 + G1 * K1;                      // [G2][K2] bf16
    float*  sB1  = (float*)(sW2 + G2 * K2);            // [G1]
    float*  sB2  = sB1 + G1;                           // [G2]
    float*  sF1w = sB2 + G2;                           // [16*H2]
    float*  sF1b = sF1w + 16 * H2DIM;                  // [16]
    float*  sF2w = sF1b + 16;                          // [16]
    float*  sF2b = sF2w + 16;                          // [1]
    __bf16* sH1  = (__bf16*)(sF2b + 1);                // [WPB][16][H1]
    __bf16* sH2  = sH1 + WPB * 16 * H1DIM;             // [WPB][16][H2]

    const int tid = threadIdx.x;

    // ---- cooperative weight staging: W1cat = [Wih1 ; Whh1]^T, bf16 -----
    for (int i = tid; i < G1 * K1; i += NTHR) {
        int g = i / K1, k = i % K1;
        float v = (k < DIN) ? Wih1[g * DIN + k] : Whh1[g * H1DIM + (k - DIN)];
        sW1[g * K1 + k] = (__bf16)v;
    }
    for (int i = tid; i < G2 * K2; i += NTHR) {
        int g = i / K2, k = i % K2;
        float v = (k < H1DIM) ? Wih2[g * H1DIM + k] : Whh2[g * H2DIM + (k - H1DIM)];
        sW2[g * K2 + k] = (__bf16)v;
    }
    for (int i = tid; i < G1; i += NTHR) sB1[i] = bih1[i] + bhh1[i];
    for (int i = tid; i < G2; i += NTHR) sB2[i] = bih2[i] + bhh2[i];
    for (int i = tid; i < 16 * H2DIM; i += NTHR) sF1w[i] = fc1w[i];
    for (int i = tid; i < 16; i += NTHR) { sF1b[i] = fc1b[i]; sF2w[i] = fc2w[i]; }
    if (tid == 0) sF2b[0] = fc2b[0];
    for (int i = tid; i < WPB * 16 * H1DIM; i += NTHR) sH1[i] = (__bf16)0.0f;
    for (int i = tid; i < WPB * 16 * H2DIM; i += NTHR) sH2[i] = (__bf16)0.0f;
    __syncthreads();

    const int wave = tid >> 5;
    const int lane = tid & 31;
    const int lcol = lane & 15;   // B/C/D column, A row
    const int hi   = lane >> 4;   // half-wave selector
    const int b0   = (blockIdx.x * WPB + wave) * 16;

    __bf16* h1 = sH1 + wave * 16 * H1DIM;   // per-wave private h1 [16][64]
    __bf16* h2 = sH2 + wave * 16 * H2DIM;   // per-wave private h2 [16][32]

    // persistent cell state in D-tile layout (lane lcol = column, VGPR e + 8*hi = row)
    v8f c1s[4], c2s[2];
#pragma unroll
    for (int i = 0; i < 4; ++i) c1s[i] = splat8(0.0f);
#pragma unroll
    for (int i = 0; i < 2; ++i) c2s[i] = splat8(0.0f);

    v16bf ah1a = load_a_lds(h1, H1DIM, lcol, 0, hi);
    v16bf ah1b = load_a_lds(h1, H1DIM, lcol, 32, hi);
    v16bf ah2  = load_a_lds(h2, H2DIM, lcol, 0, hi);

#pragma unroll 1
    for (int t = 0; t < T_STEPS; ++t) {
        if (t + 1 < T_STEPS)
            __builtin_prefetch(x + ((size_t)(b0 + lcol) * T_STEPS + (t + 1)) * DIN, 0, 0);

        v16bf ax = load_a_x(x, b0, t, lcol, hi);

        // ---------------- layer 1: 4 hidden-column groups -----------------
#pragma unroll
        for (int hc = 0; hc < 4; ++hc) {
            v8f gts[4];
#pragma unroll
            for (int gi = 0; gi < 4; ++gi) {          // i, f, g, o
                int gb = (gi * 4 + hc) * 16 + lcol;   // gate column
                v8f acc = splat8(sB1[gb]);
                acc = wmma_bf16(ax,   load_b_lds(sW1, K1, gb,  0, hi), acc);
                acc = wmma_bf16(ah1a, load_b_lds(sW1, K1, gb, 32, hi), acc);
                acc = wmma_bf16(ah1b, load_b_lds(sW1, K1, gb, 64, hi), acc);
                gts[gi] = acc;
            }
#pragma unroll
            for (int e = 0; e < 8; ++e) {
                float iv = sigm_f(gts[0][e]);
                float fv = sigm_f(gts[1][e]);
                float gv = tanh_f(gts[2][e]);
                float ov = sigm_f(gts[3][e]);
                float c  = __builtin_fmaf(fv, c1s[hc][e], iv * gv);
                c1s[hc][e] = c;
                float h  = ov * tanh_f(c);
                int m = e + hi * 8;                   // D-tile row
                h1[m * H1DIM + hc * 16 + lcol] = (__bf16)h;
            }
        }
        // reload new h1 as A fragments (same-wave LDS ordering guarantees RAW)
        ah1a = load_a_lds(h1, H1DIM, lcol, 0, hi);
        ah1b = load_a_lds(h1, H1DIM, lcol, 32, hi);

        // ---------------- layer 2: 2 hidden-column groups -----------------
#pragma unroll
        for (int hc = 0; hc < 2; ++hc) {
            v8f gts[4];
#pragma unroll
            for (int gi = 0; gi < 4; ++gi) {
                int gb = (gi * 2 + hc) * 16 + lcol;
                v8f acc = splat8(sB2[gb]);
                acc = wmma_bf16(ah1a, load_b_lds(sW2, K2, gb,  0, hi), acc);
                acc = wmma_bf16(ah1b, load_b_lds(sW2, K2, gb, 32, hi), acc);
                acc = wmma_bf16(ah2,  load_b_lds(sW2, K2, gb, 64, hi), acc);
                gts[gi] = acc;
            }
#pragma unroll
            for (int e = 0; e < 8; ++e) {
                float iv = sigm_f(gts[0][e]);
                float fv = sigm_f(gts[1][e]);
                float gv = tanh_f(gts[2][e]);
                float ov = sigm_f(gts[3][e]);
                float c  = __builtin_fmaf(fv, c2s[hc][e], iv * gv);
                c2s[hc][e] = c;
                float h  = ov * tanh_f(c);
                int m = e + hi * 8;
                h2[m * H2DIM + hc * 16 + lcol] = (__bf16)h;
            }
        }
        ah2 = load_a_lds(h2, H2DIM, lcol, 0, hi);
    }

    // ---------------- MLP head on last h2 (lanes 0..15, one row each) ------
    if (lane < 16) {
        int r = lane;
        float acc2 = sF2b[0];
#pragma unroll
        for (int u = 0; u < 16; ++u) {
            float s = sF1b[u];
#pragma unroll
            for (int j = 0; j < H2DIM; ++j)
                s = __builtin_fmaf((float)h2[r * H2DIM + j], sF1w[u * H2DIM + j], s);
            s = fmaxf(s, 0.0f);
            acc2 = __builtin_fmaf(s, sF2w[u], acc2);
        }
        out[b0 + r] = acc2;
    }
}

extern "C" void kernel_launch(void* const* d_in, const int* in_sizes, int n_in,
                              void* d_out, int out_size, void* d_ws, size_t ws_size,
                              hipStream_t stream) {
    const float* x    = (const float*)d_in[0];
    const float* Wih1 = (const float*)d_in[1];
    const float* Whh1 = (const float*)d_in[2];
    const float* bih1 = (const float*)d_in[3];
    const float* bhh1 = (const float*)d_in[4];
    const float* Wih2 = (const float*)d_in[5];
    const float* Whh2 = (const float*)d_in[6];
    const float* bih2 = (const float*)d_in[7];
    const float* bhh2 = (const float*)d_in[8];
    const float* fc1w = (const float*)d_in[9];
    const float* fc1b = (const float*)d_in[10];
    const float* fc2w = (const float*)d_in[11];
    const float* fc2b = (const float*)d_in[12];
    float* out = (float*)d_out;

    const int B = in_sizes[0] / (T_STEPS * DIN);   // 4096
    const int blocks = B / (WPB * 16);             // 32

    size_t smem = (size_t)(G1 * K1 + G2 * K2) * sizeof(__bf16)
                + (size_t)(G1 + G2 + 16 * H2DIM + 16 + 16 + 1) * sizeof(float)
                + (size_t)(WPB * 16 * H1DIM + WPB * 16 * H2DIM) * sizeof(__bf16);

    (void)hipFuncSetAttribute((const void*)lstm_fused_kernel,
                              hipFuncAttributeMaxDynamicSharedMemorySize,
                              (int)smem);

    lstm_fused_kernel<<<blocks, NTHR, smem, stream>>>(
        x, Wih1, Whh1, bih1, bhh1, Wih2, Whh2, bih2, bhh2,
        fc1w, fc1b, fc2w, fc2b, out);
}
